// HybridBackbone_61598420959601
// MI455X (gfx1250) — compile-verified, matrix operand order checked by reading, not run on hardware
//
#include <hip/hip_runtime.h>
#include <hip/hip_bf16.h>

// ---------------------------------------------------------------------------
// HybridBackbone (Mamba x6 + one banded local attention) for gfx1250 (MI455X)
// All GEMM-shaped work lowered to v_wmma_f32_16x16x32_f16 (f16 in, f32 acc).
// GEMM register-blocked 32x64 per wave (8 WMMA / K-step) to lift L2 intensity.
// ---------------------------------------------------------------------------

typedef __attribute__((ext_vector_type(16))) _Float16 v16h;
typedef __attribute__((ext_vector_type(8)))  _Float16 v8h;
typedef __attribute__((ext_vector_type(8)))  float    v8f;

#define DMODEL 512
#define SSTATE 256
#define NLAYER 6
#define LSEQ   2048
#define BATCH  4
#define NHEAD  8
#define HDIM   64
#define WINSZ  256
#define NTOK   (BATCH * LSEQ)          // 8192
#define NELT   (NTOK * DMODEL)         // 4194304

// ---------------------------------------------------------------------------
// Fragment loader (wave32). Works for:
//   A fragment:  tile is row-major M x K (row = lane&15, k pairs per half-wave)
//   B fragment:  tile is Bt = N x K row-major (col n = lane&15)
// Per ISA 7.12.2: lanes 0-15 hold k = 0..7 & 16..23, lanes 16-31 hold
// k = 8..15 & 24..31, two halves per VGPR.
// ---------------------------------------------------------------------------
__device__ __forceinline__ v16h load_frag(const _Float16* __restrict__ tile, int ld) {
  int lane = threadIdx.x & 31;
  int r    = lane & 15;
  int kb   = (lane < 16) ? 0 : 8;
  const _Float16* p = tile + (size_t)r * ld + kb;
  v8h lo = *(const v8h*)(p);        // k = kb .. kb+7
  v8h hi = *(const v8h*)(p + 16);   // k = 16+kb .. 16+kb+7
  v16h f;
#pragma unroll
  for (int i = 0; i < 8; ++i) { f[i] = lo[i]; f[8 + i] = hi[i]; }
  return f;
}

__device__ __forceinline__ v8f wmma16(v16h a, v16h b, v8f c) {
  return __builtin_amdgcn_wmma_f32_16x16x32_f16(
      /*neg_a=*/false, a, /*neg_b=*/false, b,
      /*c_mod=*/(short)0, c, /*reuse_a=*/false, /*reuse_b=*/false);
}

// ---------------------------------------------------------------------------
// Generic GEMM:  C[M x N] (f32) = A[M x K] (f16, row-major) * Bt[N x K]^T + bias
// One wave per 32x64 macro-tile: 2 A-frags x 4 B-frags -> 8 WMMAs per K-step.
// Requires M % 32 == 0, N % 64 == 0, K % 32 == 0.
// ---------------------------------------------------------------------------
__global__ void __launch_bounds__(256) gemm_f16_kernel(
    const _Float16* __restrict__ A, const _Float16* __restrict__ Bt,
    const float* __restrict__ bias, float* __restrict__ C,
    int M, int N, int K) {
  int wave   = blockIdx.x * (blockDim.x >> 5) + (threadIdx.x >> 5);
  int tilesN = N >> 6;                          // 64-wide macro-tiles
  int tiles  = (M >> 5) * tilesN;               // 32-tall macro-tiles
  if (wave >= tiles) return;                    // wave-uniform, EXEC stays full
  int tm = wave / tilesN;
  int tn = wave % tilesN;
  const _Float16* Ab0 = A  + (size_t)(tm * 32) * K;
  const _Float16* Ab1 = Ab0 + (size_t)16 * K;
  const _Float16* Bb  = Bt + (size_t)(tn * 64) * K;

  v8f acc[2][4];
#pragma unroll
  for (int i = 0; i < 2; ++i)
#pragma unroll
    for (int j = 0; j < 4; ++j) acc[i][j] = (v8f){};

  for (int k0 = 0; k0 < K; k0 += 32) {
    // prefetch next K-panel (global_prefetch_b8)
    if (k0 + 128 < K) {
      __builtin_prefetch(Ab0 + k0 + 128, 0, 1);
      __builtin_prefetch(Bb  + k0 + 128, 0, 1);
    }
    v16h a0 = load_frag(Ab0 + k0, K);
    v16h a1 = load_frag(Ab1 + k0, K);
#pragma unroll
    for (int j = 0; j < 4; ++j) {
      v16h b = load_frag(Bb + (size_t)(j * 16) * K + k0, K);
      acc[0][j] = wmma16(a0, b, acc[0][j]);
      acc[1][j] = wmma16(a1, b, acc[1][j]);
    }
  }

  int lane = threadIdx.x & 31;
  int col  = lane & 15;
  int rb   = (lane < 16) ? 0 : 8;
#pragma unroll
  for (int j = 0; j < 4; ++j) {
    int n    = tn * 64 + j * 16 + col;
    float bv = bias ? bias[n] : 0.0f;
#pragma unroll
    for (int i = 0; i < 2; ++i) {
      int m0 = tm * 32 + i * 16 + rb;
#pragma unroll
      for (int r = 0; r < 8; ++r)
        C[(size_t)(m0 + r) * N + n] = acc[i][j][r] + bv;
    }
  }
}

// ---------------------------------------------------------------------------
// Elementwise / small kernels
// ---------------------------------------------------------------------------
__global__ void init_seq_kernel(const float* __restrict__ ctx,
                                float* __restrict__ seq,
                                _Float16* __restrict__ seq16) {
  int idx = blockIdx.x * blockDim.x + threadIdx.x;
  if (idx >= NELT) return;
  float v = ctx[idx];
  seq[idx]   = v;
  seq16[idx] = (_Float16)v;
}

__global__ void zero_kernel(float* __restrict__ p, int n) {
  int idx = blockIdx.x * blockDim.x + threadIdx.x;
  if (idx < n) p[idx] = 0.0f;
}

__global__ void copy_out_kernel(const float* __restrict__ seq, float* __restrict__ out) {
  int idx = blockIdx.x * blockDim.x + threadIdx.x;
  if (idx < NELT) out[idx] = seq[idx];
}

// Convert weights to f16; Wx transposed to [layer][s][d].
__global__ void cvt_weights_kernel(const float* __restrict__ Wx,
                                   const float* __restrict__ Wqkv,
                                   const float* __restrict__ Wo,
                                   _Float16* __restrict__ WxT16,
                                   _Float16* __restrict__ Wqkv16,
                                   _Float16* __restrict__ Wo16) {
  int idx = blockIdx.x * blockDim.x + threadIdx.x;
  if (idx < NLAYER * DMODEL * SSTATE) {
    int s = idx % SSTATE;
    int d = (idx / SSTATE) % DMODEL;
    int n = idx / (SSTATE * DMODEL);
    WxT16[(size_t)n * SSTATE * DMODEL + (size_t)s * DMODEL + d] = (_Float16)Wx[idx];
  }
  if (idx < 3 * DMODEL * DMODEL) Wqkv16[idx] = (_Float16)Wqkv[idx];
  if (idx < DMODEL * DMODEL)     Wo16[idx]   = (_Float16)Wo[idx];
}

// Causal depthwise conv (k=3) + SiLU -> f16
__global__ void conv_silu_kernel(const float* __restrict__ seq,
                                 const float* __restrict__ cw,   // 3 x D (layer slice)
                                 _Float16* __restrict__ x16) {
  int idx = blockIdx.x * blockDim.x + threadIdx.x;
  if (idx >= NELT) return;
  int d = idx % DMODEL;
  int l = (idx / DMODEL) % LSEQ;
  float w0 = cw[d], w1 = cw[DMODEL + d], w2 = cw[2 * DMODEL + d];
  float x2 = seq[idx];
  float x1 = (l >= 1) ? seq[idx - DMODEL]     : 0.0f;
  float x0 = (l >= 2) ? seq[idx - 2 * DMODEL] : 0.0f;
  float x  = w0 * x0 + w1 * x1 + w2 * x2;
  float s  = x / (1.0f + __expf(-x));         // SiLU
  x16[idx] = (_Float16)s;
}

// h0[b,s] = sum_d state[b,d] * Wh[d,s]
__global__ void h0_kernel(const float* __restrict__ state,
                          const float* __restrict__ Wh,   // D x S (layer slice)
                          float* __restrict__ h0) {
  int idx = blockIdx.x * blockDim.x + threadIdx.x;
  if (idx >= BATCH * SSTATE) return;
  int s = idx % SSTATE, b = idx / SSTATE;
  float acc = 0.0f;
  for (int d = 0; d < DMODEL; ++d)
    acc += state[b * DMODEL + d] * Wh[d * SSTATE + s];
  h0[idx] = acc;
}

// hT[b,s] = a^L * h0 + sum_l u[b,l,s] * a^(L-1-l); a = sigmoid(A[s])
__global__ void __launch_bounds__(256) hT_kernel(const float* __restrict__ u,
                                                 const float* __restrict__ h0,
                                                 const float* __restrict__ Avec, // S (layer slice)
                                                 float* __restrict__ hT) {
  int bs = blockIdx.x;
  int s  = bs % SSTATE, b = bs / SSTATE;
  float a    = 1.0f / (1.0f + __expf(-Avec[s]));
  float loga = __logf(a);
  float partial = 0.0f;
  for (int l = threadIdx.x; l < LSEQ; l += 256) {
    float w = __expf((float)(LSEQ - 1 - l) * loga);
    partial += u[((size_t)b * LSEQ + l) * SSTATE + s] * w;
  }
  __shared__ float red[256];
  red[threadIdx.x] = partial;
  __syncthreads();
  for (int o = 128; o > 0; o >>= 1) {
    if (threadIdx.x < o) red[threadIdx.x] += red[threadIdx.x + o];
    __syncthreads();
  }
  if (threadIdx.x == 0)
    hT[bs] = __expf((float)LSEQ * loga) * h0[bs] + red[0];
}

// state[b,d] = sum_s hT[b,s] * Wout[s,d]
__global__ void stateout_kernel(const float* __restrict__ hT,
                                const float* __restrict__ Wout,   // S x D (layer slice)
                                float* __restrict__ state) {
  int idx = blockIdx.x * blockDim.x + threadIdx.x;
  if (idx >= BATCH * DMODEL) return;
  int d = idx % DMODEL, b = idx / DMODEL;
  float acc = 0.0f;
  for (int s = 0; s < SSTATE; ++s)
    acc += hT[b * SSTATE + s] * Wout[s * DMODEL + d];
  state[idx] = acc;
}

// seq = LayerNorm(seq + add) ; also write f16 copy.  perTok: add is (B,L,D) vs (B,D)
__global__ void __launch_bounds__(256) addnorm_kernel(
    const float* __restrict__ seqin, const float* __restrict__ add, int perTok,
    const float* __restrict__ gamma, const float* __restrict__ beta,
    float* __restrict__ seqout, _Float16* __restrict__ seq16) {
  int row = blockIdx.x;                       // b*L + l
  int b   = row / LSEQ;
  size_t rbase = (size_t)row * DMODEL;
  const float* ap = add + (perTok ? rbase : (size_t)b * DMODEL);
  int t = threadIdx.x;
  float v0 = seqin[rbase + t]       + ap[t];
  float v1 = seqin[rbase + t + 256] + ap[t + 256];
  __shared__ float red[256];
  red[t] = v0 + v1;
  __syncthreads();
  for (int o = 128; o > 0; o >>= 1) { if (t < o) red[t] += red[t + o]; __syncthreads(); }
  float mu = red[0] * (1.0f / DMODEL);
  __syncthreads();
  float d0 = v0 - mu, d1 = v1 - mu;
  red[t] = d0 * d0 + d1 * d1;
  __syncthreads();
  for (int o = 128; o > 0; o >>= 1) { if (t < o) red[t] += red[t + o]; __syncthreads(); }
  float rstd = rsqrtf(red[0] * (1.0f / DMODEL) + 1e-5f);
  float y0 = d0 * rstd * gamma[t]       + beta[t];
  float y1 = d1 * rstd * gamma[t + 256] + beta[t + 256];
  seqout[rbase + t]       = y0;
  seqout[rbase + t + 256] = y1;
  seq16[rbase + t]        = (_Float16)y0;
  seq16[rbase + t + 256]  = (_Float16)y1;
}

// Split qkv (B,L,3D) f32 -> per-head f16 layouts: q,k (B,H,L,DH), v^T (B,H,DH,L)
__global__ void qkvsplit_kernel(const float* __restrict__ qkv,
                                _Float16* __restrict__ q16,
                                _Float16* __restrict__ k16,
                                _Float16* __restrict__ v16t) {
  int idx = blockIdx.x * blockDim.x + threadIdx.x;
  if (idx >= NELT) return;
  int d = idx % DMODEL;
  int l = (idx / DMODEL) % LSEQ;
  int b = idx / (DMODEL * LSEQ);
  int h = d / HDIM, dh = d % HDIM;
  size_t base = (size_t)(b * LSEQ + l) * (3 * DMODEL);
  float qv = qkv[base + d];
  float kv = qkv[base + DMODEL + d];
  float vv = qkv[base + 2 * DMODEL + d];
  size_t hoff = ((size_t)(b * NHEAD + h) * LSEQ + l) * HDIM + dh;
  q16[hoff] = (_Float16)qv;
  k16[hoff] = (_Float16)kv;
  v16t[((size_t)(b * NHEAD + h) * HDIM + dh) * LSEQ + l] = (_Float16)vv;
}

// ---------------------------------------------------------------------------
// Banded flash attention: one wave per (b,h,16-row Q tile); 32 keys per step.
// Scores via 2x (QK^T) WMMA pairs, streaming softmax with per-lane row stats,
// P staged through LDS to re-fragment as a WMMA A operand, 4 PV WMMAs (DH=64).
// ---------------------------------------------------------------------------
__global__ void __launch_bounds__(128) attn_kernel(
    const _Float16* __restrict__ q16, const _Float16* __restrict__ k16,
    const _Float16* __restrict__ v16t, _Float16* __restrict__ o16) {
  __shared__ __align__(16) _Float16 pbuf[4][16 * 32];
  int wv   = threadIdx.x >> 5;
  int lane = threadIdx.x & 31;
  int gw   = blockIdx.x * 4 + wv;             // over B*H*(L/16) = 4096 tiles
  int nq   = LSEQ / 16;
  int qt   = gw % nq;
  int bh   = gw / nq;                          // b*H + h
  int qi0  = qt * 16;

  const _Float16* qbase = q16 + ((size_t)bh * LSEQ + qi0) * HDIM;
  v16h qa0 = load_frag(qbase, HDIM);           // k = 0..31 of head dim
  v16h qa1 = load_frag(qbase + 32, HDIM);      // k = 32..63

  v8f O0 = {}, O1 = {}, O2 = {}, O3 = {};
  float mrun[8], lrun[8];
#pragma unroll
  for (int r = 0; r < 8; ++r) { mrun[r] = -3.0e38f; lrun[r] = 0.0f; }

  int lo = qi0 - WINSZ; if (lo < 0) lo = 0; lo &= ~31;
  int hi = qi0 + 16 + WINSZ; hi = (hi + 31) & ~31; if (hi > LSEQ) hi = LSEQ;

  int col     = lane & 15;
  int rowbase = (lane < 16) ? 0 : 8;
  _Float16* pl = pbuf[wv];

  for (int kt = lo; kt < hi; kt += 32) {
    // ---- scores: S(16x32) = Q(16x64) @ K^T ----
    const _Float16* kb = k16 + ((size_t)bh * LSEQ + kt) * HDIM;
    v16h kb0a = load_frag(kb, HDIM);
    v16h kb0b = load_frag(kb + 32, HDIM);
    v16h kb1a = load_frag(kb + 16 * HDIM, HDIM);
    v16h kb1b = load_frag(kb + 16 * HDIM + 32, HDIM);
    v8f s0 = {}; s0 = wmma16(qa0, kb0a, s0); s0 = wmma16(qa1, kb0b, s0);
    v8f s1 = {}; s1 = wmma16(qa0, kb1a, s1); s1 = wmma16(qa1, kb1b, s1);

    // ---- scale + band mask ----
    float sc0[8], sc1[8];
#pragma unroll
    for (int r = 0; r < 8; ++r) {
      int qi = qi0 + rowbase + r;
      int d0 = qi - (kt + col);      if (d0 < 0) d0 = -d0;
      int d1 = qi - (kt + 16 + col); if (d1 < 0) d1 = -d1;
      sc0[r] = (d0 <= WINSZ) ? s0[r] * 0.125f : -3.0e38f;
      sc1[r] = (d1 <= WINSZ) ? s1[r] * 0.125f : -3.0e38f;
    }

    // ---- row max across the 16 lanes of each half-wave ----
    float rm[8];
#pragma unroll
    for (int r = 0; r < 8; ++r) rm[r] = fmaxf(sc0[r], sc1[r]);
    for (int off = 1; off < 16; off <<= 1) {
#pragma unroll
      for (int r = 0; r < 8; ++r) rm[r] = fmaxf(rm[r], __shfl_xor(rm[r], off, 32));
    }

    // ---- streaming softmax update ----
    float scl[8], p0[8], p1[8], rs[8];
#pragma unroll
    for (int r = 0; r < 8; ++r) {
      float mn = fmaxf(mrun[r], rm[r]);
      scl[r]  = __expf(mrun[r] - mn);
      mrun[r] = mn;
      p0[r] = (sc0[r] < -1.0e37f) ? 0.0f : __expf(sc0[r] - mn);
      p1[r] = (sc1[r] < -1.0e37f) ? 0.0f : __expf(sc1[r] - mn);
      rs[r] = p0[r] + p1[r];
    }
    for (int off = 1; off < 16; off <<= 1) {
#pragma unroll
      for (int r = 0; r < 8; ++r) rs[r] += __shfl_xor(rs[r], off, 32);
    }
#pragma unroll
    for (int r = 0; r < 8; ++r) {
      lrun[r] = lrun[r] * scl[r] + rs[r];
      O0[r] *= scl[r]; O1[r] *= scl[r]; O2[r] *= scl[r]; O3[r] *= scl[r];
    }

    // ---- stage P (16x32 f16) through LDS to build a WMMA A fragment ----
#pragma unroll
    for (int r = 0; r < 8; ++r) {
      int row = rowbase + r;
      pl[row * 32 + col]      = (_Float16)p0[r];
      pl[row * 32 + 16 + col] = (_Float16)p1[r];
    }
    asm volatile("s_wait_dscnt 0" ::: "memory");   // stores visible before re-read
    v16h pa = load_frag(pl, 32);

    // ---- O(16x64) += P(16x32) @ V(32x64), V stored transposed (DH x L) ----
    const _Float16* vb = v16t + (size_t)bh * HDIM * LSEQ + kt;
    v16h vb0 = load_frag(vb, LSEQ);
    v16h vb1 = load_frag(vb + (size_t)16 * LSEQ, LSEQ);
    v16h vb2 = load_frag(vb + (size_t)32 * LSEQ, LSEQ);
    v16h vb3 = load_frag(vb + (size_t)48 * LSEQ, LSEQ);
    O0 = wmma16(pa, vb0, O0);
    O1 = wmma16(pa, vb1, O1);
    O2 = wmma16(pa, vb2, O2);
    O3 = wmma16(pa, vb3, O3);
    asm volatile("" ::: "memory");                 // keep next-iter P stores below
  }

  // ---- normalize and write f16 output (feeds Wo GEMM) ----
  int h = bh % NHEAD;
  int b = bh / NHEAD;
#pragma unroll
  for (int r = 0; r < 8; ++r) {
    float inv = 1.0f / lrun[r];
    int row = qi0 + rowbase + r;
    size_t o = (size_t)(b * LSEQ + row) * DMODEL + h * HDIM;
    o16[o + 0  + col] = (_Float16)(O0[r] * inv);
    o16[o + 16 + col] = (_Float16)(O1[r] * inv);
    o16[o + 32 + col] = (_Float16)(O2[r] * inv);
    o16[o + 48 + col] = (_Float16)(O3[r] * inv);
  }
}

// ---------------------------------------------------------------------------
// Host orchestration
// ---------------------------------------------------------------------------
extern "C" void kernel_launch(void* const* d_in, const int* in_sizes, int n_in,
                              void* d_out, int out_size, void* d_ws, size_t ws_size,
                              hipStream_t stream) {
  (void)in_sizes; (void)n_in; (void)out_size; (void)ws_size;

  const float* ctx     = (const float*)d_in[0];
  const float* conv_w  = (const float*)d_in[1];
  const float* Wx      = (const float*)d_in[2];
  const float* Wh      = (const float*)d_in[3];
  const float* Wout    = (const float*)d_in[4];
  const float* Avec    = (const float*)d_in[5];
  const float* gamma_m = (const float*)d_in[6];
  const float* beta_m  = (const float*)d_in[7];
  const float* gamma_a = (const float*)d_in[8];
  const float* beta_a  = (const float*)d_in[9];
  const float* Wqkv    = (const float*)d_in[10];
  const float* bqkv    = (const float*)d_in[11];
  const float* Wo      = (const float*)d_in[12];
  const float* bo      = (const float*)d_in[13];
  float* out = (float*)d_out;

  char*  ws  = (char*)d_ws;
  size_t off = 0;
  auto alloc = [&](size_t bytes) -> void* {
    void* p = ws + off;
    off += (bytes + 255) & ~(size_t)255;
    return p;
  };

  float*     seq    = (float*)alloc((size_t)NELT * 4);
  _Float16*  seq16  = (_Float16*)alloc((size_t)NELT * 2);
  _Float16*  x16    = (_Float16*)alloc((size_t)NELT * 2);     // also attn output
  float*     u      = (float*)alloc((size_t)NTOK * SSTATE * 4);
  float*     state  = (float*)alloc((size_t)BATCH * DMODEL * 4);
  float*     h0     = (float*)alloc((size_t)BATCH * SSTATE * 4);
  float*     hT     = (float*)alloc((size_t)BATCH * SSTATE * 4);
  _Float16*  WxT16  = (_Float16*)alloc((size_t)NLAYER * SSTATE * DMODEL * 2);
  _Float16*  Wqkv16 = (_Float16*)alloc((size_t)3 * DMODEL * DMODEL * 2);
  _Float16*  Wo16   = (_Float16*)alloc((size_t)DMODEL * DMODEL * 2);
  float*     qkv    = (float*)alloc((size_t)NTOK * 3 * DMODEL * 4);
  _Float16*  q16    = (_Float16*)alloc((size_t)NELT * 2);
  _Float16*  k16    = (_Float16*)alloc((size_t)NELT * 2);
  _Float16*  v16t   = (_Float16*)alloc((size_t)NELT * 2);
  float*     attnbuf = qkv;                                   // reuse (qkv dead after split)
  _Float16*  o16     = x16;                                   // reuse (x16 dead during attn)

  const int TB = 256;
  init_seq_kernel<<<NELT / TB, TB, 0, stream>>>(ctx, seq, seq16);
  zero_kernel<<<(BATCH * DMODEL + TB - 1) / TB, TB, 0, stream>>>(state, BATCH * DMODEL);
  cvt_weights_kernel<<<(NLAYER * DMODEL * SSTATE + TB - 1) / TB, TB, 0, stream>>>(
      Wx, Wqkv, Wo, WxT16, Wqkv16, Wo16);

  for (int i = 0; i < NLAYER; ++i) {
    conv_silu_kernel<<<NELT / TB, TB, 0, stream>>>(seq, conv_w + (size_t)i * 3 * DMODEL, x16);

    int tiles = (NTOK / 32) * (SSTATE / 64);
    gemm_f16_kernel<<<(tiles + 7) / 8, TB, 0, stream>>>(
        x16, WxT16 + (size_t)i * SSTATE * DMODEL, nullptr, u, NTOK, SSTATE, DMODEL);

    h0_kernel<<<(BATCH * SSTATE + TB - 1) / TB, TB, 0, stream>>>(
        state, Wh + (size_t)i * DMODEL * SSTATE, h0);
    hT_kernel<<<BATCH * SSTATE, TB, 0, stream>>>(u, h0, Avec + (size_t)i * SSTATE, hT);
    stateout_kernel<<<(BATCH * DMODEL + TB - 1) / TB, TB, 0, stream>>>(
        hT, Wout + (size_t)i * SSTATE * DMODEL, state);

    addnorm_kernel<<<NTOK, TB, 0, stream>>>(seq, state, /*perTok=*/0,
                                            gamma_m, beta_m, seq, seq16);

    if ((i + 1) % 4 == 0) {
      tiles = (NTOK / 32) * ((3 * DMODEL) / 64);
      gemm_f16_kernel<<<(tiles + 7) / 8, TB, 0, stream>>>(
          seq16, Wqkv16, bqkv, qkv, NTOK, 3 * DMODEL, DMODEL);
      qkvsplit_kernel<<<NELT / TB, TB, 0, stream>>>(qkv, q16, k16, v16t);
      attn_kernel<<<(BATCH * NHEAD * (LSEQ / 16)) / 4, 128, 0, stream>>>(q16, k16, v16t, o16);
      tiles = (NTOK / 32) * (DMODEL / 64);
      gemm_f16_kernel<<<(tiles + 7) / 8, TB, 0, stream>>>(
          o16, Wo16, bo, attnbuf, NTOK, DMODEL, DMODEL);
      addnorm_kernel<<<NTOK, TB, 0, stream>>>(seq, attnbuf, /*perTok=*/1,
                                              gamma_a, beta_a, seq, seq16);
    }
  }

  copy_out_kernel<<<NELT / TB, TB, 0, stream>>>(seq, out);
}
